// MultiHeadDiffAttention_19353122635966
// MI455X (gfx1250) — matrix operand order repaired and checked by reading, not compiled, not run on hardware
//
#include <hip/hip_runtime.h>

// ---------------------------------------------------------------------------
// DiffAttention for MI455X (gfx1250), all GEMMs on v_wmma_f32_16x16x32_bf16.
// B=2, T=2048, C=1024, H=16, hs=64. ~137 GFLOP; working set ~90MB (L2-resident).
// Attention: two softmax streams split across waves to keep each wave under
// 256 VGPRs (no s_set_vgpr_msb churn in the WMMA loop).
// ---------------------------------------------------------------------------

#define B_   2
#define T_   2048
#define C_   1024
#define H_   16
#define HS_  64
#define M_   (B_ * T_)                 // 4096 tokens
#define LAMBDA_INIT_F 0.55605820f      // 0.8 - 0.6*exp(-0.3*(4-1))

typedef __attribute__((ext_vector_type(16))) __bf16 v16bf;
typedef __attribute__((ext_vector_type(8)))  __bf16 v8bf;
typedef __attribute__((ext_vector_type(8)))  float  v8f;

__device__ __forceinline__ v8f wmma_bf16(v16bf a, v16bf b, v8f c) {
  // (neg_a, A, neg_b, B, c_mod, C, reuse_a, reuse_b)
  return __builtin_amdgcn_wmma_f32_16x16x32_bf16(false, a, false, b, (short)0, c,
                                                 false, false);
}

// A-fragment (16x32 bf16, contiguous k order, same as load_bfrag): lane m = lane%16, hi = lane>>4.
//   elements 0..7  = row[k0 + hi*16 + i]
//   elements 8..15 = row[k0 + hi*16 + 8 + i]
__device__ __forceinline__ v16bf load_afrag(const __bf16* rowptr, int k0, int hi) {
  v8bf lo = *reinterpret_cast<const v8bf*>(rowptr + k0 + hi * 16);
  v8bf hh = *reinterpret_cast<const v8bf*>(rowptr + k0 + hi * 16 + 8);
  v16bf r;
#pragma unroll
  for (int i = 0; i < 8; ++i) { r[i] = lo[i]; r[i + 8] = hh[i]; }
  return r;
}

// B-fragment (32x16 bf16): lane n = lane%16, hi = lane>>4, element e = B[k0+hi*16+e][n].
// Caller passes pointer to 16 contiguous bf16 along K (32B aligned).
__device__ __forceinline__ v16bf load_bfrag(const __bf16* p) {
  return *reinterpret_cast<const v16bf*>(p);
}

// reductions across the 16-lane half (wave32, C-layout columns live in lanes)
__device__ __forceinline__ float half_reduce_max(float v) {
  v = fmaxf(v, __shfl_xor(v, 1, 32));
  v = fmaxf(v, __shfl_xor(v, 2, 32));
  v = fmaxf(v, __shfl_xor(v, 4, 32));
  v = fmaxf(v, __shfl_xor(v, 8, 32));
  return v;
}
__device__ __forceinline__ float half_reduce_sum(float v) {
  v += __shfl_xor(v, 1, 32);
  v += __shfl_xor(v, 2, 32);
  v += __shfl_xor(v, 4, 32);
  v += __shfl_xor(v, 8, 32);
  return v;
}

// ---------------------------------------------------------------------------
// elementwise fp32 -> bf16
__global__ void cvt_bf16_kernel(const float* __restrict__ in,
                                __bf16* __restrict__ out, int n) {
  int i = blockIdx.x * blockDim.x + threadIdx.x;
  if (i < n) out[i] = (__bf16)in[i];
}

// fp32 [K][N] -> bf16 [N][K] (transpose so WMMA B-frags are contiguous along K)
__global__ void cvt_T_bf16_kernel(const float* __restrict__ in,
                                  __bf16* __restrict__ out, int Kd, int Nd) {
  int i = blockIdx.x * blockDim.x + threadIdx.x;
  if (i < Kd * Nd) {
    int n = i / Kd, k = i % Kd;
    out[i] = (__bf16)in[(size_t)k * Nd + n];
  }
}

// lam[h] = exp(<lq1[h],lk1[h]>) - exp(<lq2[h],lk2[h]>) + lambda_init
__global__ __launch_bounds__(512) void lambda_kernel(
    const float* __restrict__ lq1, const float* __restrict__ lk1,
    const float* __restrict__ lq2, const float* __restrict__ lk2,
    float* __restrict__ lam) {
  int h = threadIdx.x >> 5, lane = threadIdx.x & 31;
  int i0 = h * HS_ + lane, i1 = i0 + 32;
  float d1 = lq1[i0] * lk1[i0] + lq1[i1] * lk1[i1];
  float d2 = lq2[i0] * lk2[i0] + lq2[i1] * lk2[i1];
#pragma unroll
  for (int m = 1; m < 32; m <<= 1) {
    d1 += __shfl_xor(d1, m, 32);
    d2 += __shfl_xor(d2, m, 32);
  }
  if (lane == 0) lam[h] = __expf(d1) - __expf(d2) + LAMBDA_INIT_F;
}

// ---------------------------------------------------------------------------
// GEMM: C[M][N] = A[M][K] (bf16 row-major) * BT[N][K]^T (bf16, pre-transposed).
// Workgroup = 8 waves = 64(M) x 128(N) tile; each wave does 16x64 via 4 accs.
// MODE 0: fp32 row-major out. MODE 1: bf16 -> q/k layout [B][H][T][64].
// MODE 2: bf16 -> vT layout [B][H][128][T].
template <int MODE>
__global__ __launch_bounds__(256) void gemm_bf16_kernel(
    const __bf16* __restrict__ A, const __bf16* __restrict__ BT,
    void* __restrict__ Cout, int N, int K) {
  const int lane = threadIdx.x & 31;
  const int wave = threadIdx.x >> 5;
  const int col  = lane & 15;
  const int hi   = lane >> 4;
  const int m0 = blockIdx.x * 64 + (wave & 3) * 16;
  const int n0 = blockIdx.y * 128 + (wave >> 2) * 64;

  v8f acc[4] = {};
  const __bf16* arow = A + (size_t)(m0 + col) * K;

  for (int k0 = 0; k0 < K; k0 += 32) {
    v16bf af = load_afrag(arow, k0, hi);
#pragma unroll
    for (int nt = 0; nt < 4; ++nt) {
      v16bf bf = load_bfrag(BT + (size_t)(n0 + nt * 16 + col) * K + k0 + hi * 16);
      acc[nt] = wmma_bf16(af, bf, acc[nt]);
    }
  }

#pragma unroll
  for (int nt = 0; nt < 4; ++nt) {
#pragma unroll
    for (int r = 0; r < 8; ++r) {
      const int m = m0 + r + hi * 8;          // C-layout: vgpr r -> row r (+8 hi)
      const int n = n0 + nt * 16 + col;       //           lane%16 -> column
      const float v = acc[nt][r];
      if (MODE == 0) {
        ((float*)Cout)[(size_t)m * N + n] = v;
      } else if (MODE == 1) {
        const int b = m / T_, t = m % T_, h = n / HS_, d = n % HS_;
        ((__bf16*)Cout)[(((size_t)(b * H_ + h)) * T_ + t) * HS_ + d] = (__bf16)v;
      } else {
        const int b = m / T_, t = m % T_, h = n / (2 * HS_), c = n % (2 * HS_);
        ((__bf16*)Cout)[(((size_t)(b * H_ + h)) * (2 * HS_) + c) * T_ + t] = (__bf16)v;
      }
    }
  }
}

// ---------------------------------------------------------------------------
// Two-stream causal flash attention + lambda combine + per-head LayerNorm.
// Block = 8 waves = 64 queries of one (b,h).
//   waves 0..3: stream 1 (q1*k1^T), query tiles 0..3
//   waves 4..7: stream 2 (q2*k2^T), query tiles 0..3
// Each wave: 8 v8f accumulators (64 VGPRs) + q frags (16) + m/l (16) -> no
// VGPR-MSB addressing needed. Streams combine through LDS at the end.
__global__ __launch_bounds__(256) void diff_attn_kernel(
    const __bf16* __restrict__ q1g, const __bf16* __restrict__ q2g,
    const __bf16* __restrict__ k1g, const __bf16* __restrict__ k2g,
    const __bf16* __restrict__ vtg, const float* __restrict__ lam_p,
    __bf16* __restrict__ ybf) {
  __shared__ __align__(64) __bf16 plds[8][16 * 32];    // per-wave P tiles (8 KB)
  __shared__ __align__(64) float  clds[4][16 * 128];   // stream-2 results (32 KB)

  const int lane  = threadIdx.x & 31;
  const int wave  = threadIdx.x >> 5;
  const int col   = lane & 15;
  const int hi    = lane >> 4;
  const int qtile = wave & 3;
  const int sidx  = wave >> 2;                         // 0: stream1, 1: stream2

  const int qb = blockIdx.x & 31;                      // T/64 = 32
  const int h  = (blockIdx.x >> 5) & 15;
  const int b  = blockIdx.x >> 9;
  const int qrow0 = qb * 64 + qtile * 16;

  const size_t bh = (size_t)(b * H_ + h);
  const __bf16* qp  = (sidx ? q2g : q1g) + bh * T_ * HS_;
  const __bf16* kp  = (sidx ? k2g : k1g) + bh * T_ * HS_;
  const __bf16* vtp = vtg + bh * (2 * HS_) * T_;

  const __bf16* qrow = qp + (size_t)(qrow0 + col) * HS_;
  v16bf qf0 = load_afrag(qrow, 0, hi);
  v16bf qf1 = load_afrag(qrow, 32, hi);

  v8f y[8] = {};                                       // 16 queries x 128 chans
  float mrow[8], lrow[8];
#pragma unroll
  for (int r = 0; r < 8; ++r) { mrow[r] = -1e30f; lrow[r] = 0.f; }

  const int nkb = (qrow0 + 47) >> 5;                   // causal: keys 0..qrow0+15
  for (int kb32 = 0; kb32 < nkb; ++kb32) {
    const int kb = kb32 << 5;

    // S = q * k^T for 16 queries x 32 keys (K = hs = 64 -> 2 WMMAs per tile)
    v8f st0 = {}, st1 = {};
    {
      const __bf16* kr0 = kp + (size_t)(kb + col) * HS_;
      st0 = wmma_bf16(qf0, load_bfrag(kr0 + hi * 16), st0);
      st0 = wmma_bf16(qf1, load_bfrag(kr0 + 32 + hi * 16), st0);
      const __bf16* kr1 = kp + (size_t)(kb + 16 + col) * HS_;
      st1 = wmma_bf16(qf0, load_bfrag(kr1 + hi * 16), st1);
      st1 = wmma_bf16(qf1, load_bfrag(kr1 + 32 + hi * 16), st1);
    }
    const float scale = 0.125f;                        // 1/sqrt(64)
#pragma unroll
    for (int r = 0; r < 8; ++r) {
      const int row = qrow0 + r + hi * 8;
      float s0 = st0[r] * scale, s1 = st1[r] * scale;
      if (kb + col > row)      s0 = -1e30f;            // causal mask
      if (kb + 16 + col > row) s1 = -1e30f;
      const float mn = fmaxf(mrow[r], half_reduce_max(fmaxf(s0, s1)));
      const float alpha = __expf(mrow[r] - mn);
      mrow[r] = mn;
      const float p0 = __expf(s0 - mn);
      const float p1 = __expf(s1 - mn);
      lrow[r] = lrow[r] * alpha + half_reduce_sum(p0 + p1);
#pragma unroll
      for (int nt = 0; nt < 8; ++nt) y[nt][r] *= alpha;
      // P tile row-major [16][32] bf16, wave-private LDS
      plds[wave][(r + hi * 8) * 32 + col]      = (__bf16)p0;
      plds[wave][(r + hi * 8) * 32 + 16 + col] = (__bf16)p1;
    }

    // wave-private LDS: DS ops are in-order per wave; wait + compiler barrier
    asm volatile("s_wait_dscnt 0x0" ::: "memory");

    v16bf pf = load_afrag(&plds[wave][col * 32], 0, hi);
#pragma unroll
    for (int nt = 0; nt < 8; ++nt) {                   // Y += P * V (K = 32 keys)
      v16bf vf = load_bfrag(vtp + (size_t)(nt * 16 + col) * T_ + kb + hi * 16);
      y[nt] = wmma_bf16(pf, vf, y[nt]);
    }
  }

  // normalize by softmax denominator; stream 2 parks results in LDS
#pragma unroll
  for (int r = 0; r < 8; ++r) {
    const float inv = 1.f / lrow[r];
    const int row16 = r + hi * 8;
#pragma unroll
    for (int nt = 0; nt < 8; ++nt) {
      const float v = y[nt][r] * inv;
      if (sidx) clds[qtile][row16 * 128 + nt * 16 + col] = v;
      else      y[nt][r] = v;
    }
  }
  __syncthreads();

  if (sidx == 0) {
    const float lam = lam_p[h];
#pragma unroll
    for (int r = 0; r < 8; ++r) {
      const int row16 = r + hi * 8;
      float vals[8];
      float sum = 0.f;
#pragma unroll
      for (int nt = 0; nt < 8; ++nt) {
        const float v = y[nt][r] - lam * clds[qtile][row16 * 128 + nt * 16 + col];
        vals[nt] = v;
        sum += v;
      }
      const float mu = half_reduce_sum(sum) * (1.f / 128.f);
      float vsum = 0.f;
#pragma unroll
      for (int nt = 0; nt < 8; ++nt) { const float d = vals[nt] - mu; vsum += d * d; }
      const float rstd = rsqrtf(half_reduce_sum(vsum) * (1.f / 128.f) + 1e-5f);
      const int row = qrow0 + row16;
      __bf16* outp = ybf + ((size_t)(b * T_ + row)) * (2 * C_) + h * (2 * HS_);
#pragma unroll
      for (int nt = 0; nt < 8; ++nt)
        outp[nt * 16 + col] = (__bf16)((vals[nt] - mu) * rstd * (1.f - LAMBDA_INIT_F));
    }
  }
}

// ---------------------------------------------------------------------------
extern "C" void kernel_launch(void* const* d_in, const int* in_sizes, int n_in,
                              void* d_out, int out_size, void* d_ws, size_t ws_size,
                              hipStream_t stream) {
  const float* x   = (const float*)d_in[0];
  const float* q1w = (const float*)d_in[1];
  const float* q2w = (const float*)d_in[2];
  const float* k1w = (const float*)d_in[3];
  const float* k2w = (const float*)d_in[4];
  const float* vw  = (const float*)d_in[5];
  const float* cw  = (const float*)d_in[6];
  const float* lq1 = (const float*)d_in[7];
  const float* lk1 = (const float*)d_in[8];
  const float* lq2 = (const float*)d_in[9];
  const float* lk2 = (const float*)d_in[10];

  // workspace carve-up (bf16 unless noted); total ~92.3 MB
  char* ws = (char*)d_ws;
  __bf16* xb   = (__bf16*)ws;                       // [4096][1024]
  __bf16* wq1T = xb + (size_t)M_ * C_;              // [1024][1024] each
  __bf16* wq2T = wq1T + (size_t)C_ * C_;
  __bf16* wk1T = wq2T + (size_t)C_ * C_;
  __bf16* wk2T = wk1T + (size_t)C_ * C_;
  __bf16* wvT  = wk2T + (size_t)C_ * C_;            // [2048][1024]
  __bf16* wcT  = wvT + (size_t)2 * C_ * C_;         // [1024][2048]
  __bf16* q1b  = wcT + (size_t)2 * C_ * C_;         // [B][H][T][64] each
  __bf16* q2b  = q1b + (size_t)B_ * H_ * T_ * HS_;
  __bf16* k1b  = q2b + (size_t)B_ * H_ * T_ * HS_;
  __bf16* k2b  = k1b + (size_t)B_ * H_ * T_ * HS_;
  __bf16* vTb  = k2b + (size_t)B_ * H_ * T_ * HS_;  // [B][H][128][T]
  __bf16* yb   = vTb + (size_t)B_ * H_ * 2 * HS_ * T_;  // [B][T][2048]
  float*  lam  = (float*)(yb + (size_t)B_ * T_ * 2 * C_);

  const int thr = 256;
  cvt_bf16_kernel<<<(M_ * C_) / thr, thr, 0, stream>>>(x, xb, M_ * C_);
  cvt_T_bf16_kernel<<<(C_ * C_) / thr, thr, 0, stream>>>(q1w, wq1T, C_, C_);
  cvt_T_bf16_kernel<<<(C_ * C_) / thr, thr, 0, stream>>>(q2w, wq2T, C_, C_);
  cvt_T_bf16_kernel<<<(C_ * C_) / thr, thr, 0, stream>>>(k1w, wk1T, C_, C_);
  cvt_T_bf16_kernel<<<(C_ * C_) / thr, thr, 0, stream>>>(k2w, wk2T, C_, C_);
  cvt_T_bf16_kernel<<<(2 * C_ * C_) / thr, thr, 0, stream>>>(vw, wvT, C_, 2 * C_);
  cvt_T_bf16_kernel<<<(2 * C_ * C_) / thr, thr, 0, stream>>>(cw, wcT, 2 * C_, C_);
  lambda_kernel<<<1, 512, 0, stream>>>(lq1, lk1, lq2, lk2, lam);

  // projections (M=4096): q/k (N=1024) into head layout; v (N=2048) transposed
  gemm_bf16_kernel<1><<<dim3(M_ / 64, 1024 / 128), thr, 0, stream>>>(xb, wq1T, q1b, 1024, 1024);
  gemm_bf16_kernel<1><<<dim3(M_ / 64, 1024 / 128), thr, 0, stream>>>(xb, wq2T, q2b, 1024, 1024);
  gemm_bf16_kernel<1><<<dim3(M_ / 64, 1024 / 128), thr, 0, stream>>>(xb, wk1T, k1b, 1024, 1024);
  gemm_bf16_kernel<1><<<dim3(M_ / 64, 1024 / 128), thr, 0, stream>>>(xb, wk2T, k2b, 1024, 1024);
  gemm_bf16_kernel<2><<<dim3(M_ / 64, 2048 / 128), thr, 0, stream>>>(xb, wvT, vTb, 2048, 1024);

  // attention: B*H*(T/64) workgroups of 8 waves (2 streams x 4 query tiles)
  diff_attn_kernel<<<B_ * H_ * (T_ / 64), 256, 0, stream>>>(q1b, q2b, k1b, k2b, vTb, lam, yb);

  // output projection: [4096][2048] x [2048][1024] -> fp32 d_out
  gemm_bf16_kernel<0><<<dim3(M_ / 64, 1024 / 128), thr, 0, stream>>>(yb, wcT, d_out, 1024, 2048);
}
